// DiscoBertModelGlove2ClassStackOnly_21809843929910
// MI455X (gfx1250) — compile-verified, matrix operand order checked by reading, not run on hardware
//
#include <hip/hip_runtime.h>
#include <hip/hip_bf16.h>
#include <stdint.h>

typedef __attribute__((ext_vector_type(16))) __bf16 v16bf;
typedef __attribute__((ext_vector_type(8)))  float  v8f;
typedef __attribute__((ext_vector_type(4)))  unsigned int v4u;
typedef __attribute__((ext_vector_type(8)))  int v8i_;
typedef __attribute__((ext_vector_type(4)))  int v4i_;

#define HDIM 100
#define GDIM 400      // 4*H
#define NHD  200      // 2*H
#define NEDU 256
#define TT   32
#define EDIM 300
#define KP0  320      // E padded to mult of 32
#define KP12 224      // 2H padded
#define KPH  128      // H padded
#define MROWS (NEDU*TT)   // 8192

__device__ __forceinline__ unsigned short f2bf(float f) {
  uint32_t u = __float_as_uint(f);
  uint32_t r = u + 0x7FFFu + ((u >> 16) & 1u);
  return (unsigned short)(r >> 16);
}
__device__ __forceinline__ float sigmoidf_(float x){ return 1.0f/(1.0f+__expf(-x)); }

// ---- TDM: stage Whh (400 rows x 256 B) into LDS via tensor_load_to_lds -----
// D# per CDNA5 ISA: 2D tile, data_size=8B, dim0=32 elems,
// 400 rows, contiguous (stride == dim0). Issued by one wave; TENSORcnt waited.
__device__ __forceinline__ void tdm_load_whh(const unsigned short* gsrc, unsigned lds_off){
  unsigned long long ga = (unsigned long long)(uintptr_t)gsrc;
  v4u g0;
  g0.x = 1u;                                              // count=1 (valid), user mode
  g0.y = lds_off;                                         // lds_addr (bytes)
  g0.z = (unsigned)(ga & 0xFFFFFFFFu);                    // global_addr[31:0]
  g0.w = (unsigned)((ga >> 32) & 0x1FFFFFFu) | (2u << 30);// global_addr[56:32] | type=2
  v8i_ g1;
  g1[0] = (3 << 16);      // workgroup_mask=0, data_size=3 (8B), no barrier/iter/pad
  g1[1] = (32 << 16);     // tensor_dim0[15:0]=32 (8B elems per row)
  g1[2] = (400 << 16);    // tensor_dim0 hi=0 | tensor_dim1[15:0]=400 rows
  g1[3] = (32 << 16);     // tensor_dim1 hi=0 | tile_dim0=32
  g1[4] = 400;            // tile_dim1=400, tile_dim2=0
  g1[5] = 32;             // tensor_dim0_stride[31:0]=32 (contiguous rows)
  g1[6] = 0;              // stride0 hi | tensor_dim1_stride lo
  g1[7] = 0;
  v4i_ gz = {0,0,0,0};
#if __clang_major__ >= 23
  v8i_ gz8 = {0,0,0,0,0,0,0,0};
  __builtin_amdgcn_tensor_load_to_lds(g0, g1, gz, gz, gz8, 0);
#else
  __builtin_amdgcn_tensor_load_to_lds(g0, g1, gz, gz, 0);
#endif
}

// ---------------- embedding gather + bf16 convert (pad K to 320) -------------
__global__ void k_embed(const int* __restrict__ tokens, const float* __restrict__ emb,
                        unsigned short* __restrict__ we){
  int row = blockIdx.x;          // b*32 + t
  int col = threadIdx.x;         // 0..319
  int tok = tokens[row];
  float v = (col < EDIM) ? emb[(size_t)tok*EDIM + col] : 0.0f;
  we[(size_t)row*KP0 + col] = f2bf(v);
}

// ---------------- fp32 (R x K) -> bf16 (R x Kp) weight convert ---------------
__global__ void k_cvt(const float* __restrict__ src, unsigned short* __restrict__ dst,
                      int K, int Kp){
  int r = blockIdx.x, c = threadIdx.x;   // blockDim == Kp
  float v = (c < K) ? src[(size_t)r*K + c] : 0.0f;
  dst[(size_t)r*Kp + c] = f2bf(v);
}

// ---------------- zero the K-pad columns of the inter-layer buffer -----------
__global__ void k_zpad(unsigned short* __restrict__ inb){
  int idx = blockIdx.x*blockDim.x + threadIdx.x;
  int npad = KP12 - NHD;
  if (idx >= MROWS*npad) return;
  int r = idx / npad, c = NHD + idx % npad;
  inb[(size_t)r*KP12 + c] = 0;
}

// ---------------- big input-projection GEMM: xproj = X @ Wih^T ---------------
__global__ void k_gemm_xproj(const unsigned short* __restrict__ A,
                             const unsigned short* __restrict__ Ball,
                             float* __restrict__ Call, int Kp){
  int dir = blockIdx.y;
  const unsigned short* B = Ball + (size_t)dir*GDIM*Kp;
  float* C = Call + (size_t)dir*MROWS*GDIM;
  int wave = threadIdx.x >> 5;
  int lane = threadIdx.x & 31;
  int tile = blockIdx.x*8 + wave;
  const int TM = MROWS/16;                 // 512 tiles in M
  int mt = tile % TM;
  int nt = tile / TM;                      // 0..24
  if (nt >= GDIM/16) return;
  int m0 = mt*16, n0 = nt*16;
  int lrow = lane & 15;
  int ksel = (lane >> 4) << 4;             // 0 or 16
  v8f acc = {};
  for (int k0 = 0; k0 < Kp; k0 += 32) {
    v16bf a = *reinterpret_cast<const v16bf*>(A + (size_t)(m0+lrow)*Kp + k0 + ksel);
    v16bf b = *reinterpret_cast<const v16bf*>(B + (size_t)(n0+lrow)*Kp + k0 + ksel);
    acc = __builtin_amdgcn_wmma_f32_16x16x32_bf16(false, a, false, b, (short)0, acc, false, false);
  }
  int col = n0 + (lane & 15);
  int rb  = m0 + ((lane >> 4) << 3);
  #pragma unroll
  for (int r = 0; r < 8; ++r)
    C[(size_t)(rb + r)*GDIM + col] = acc[r];
}

// ---------------- recurrent LSTM scan: one persistent WG per direction -------
// LDS: c fp32 (256x100) + h bf16 (256x128) + Whh bf16 (400x128, TDM-staged)
//    = 102400 + 65536 + 102400 = 270336 B
__global__ void k_scan(const float* __restrict__ xproj2,
                       const unsigned short* __restrict__ Whh2,
                       const float* __restrict__ bias2,
                       const int* __restrict__ lengths,
                       float* __restrict__ outs, unsigned short* __restrict__ nextin,
                       float* __restrict__ gates2){
  extern __shared__ char smem[];
  float* cF = (float*)smem;                                           // 256*100 f32
  unsigned short* hB = (unsigned short*)(smem + NEDU*HDIM*4);         // 256*128 bf16
  unsigned short* hW = (unsigned short*)(smem + NEDU*HDIM*4 + NEDU*KPH*2); // 400*128 bf16
  int dir = blockIdx.x;                                   // 0 fwd, 1 bwd
  const float*          xproj = xproj2 + (size_t)dir*MROWS*GDIM;
  const unsigned short* Whh   = Whh2   + (size_t)dir*GDIM*KPH;
  const float*          bias  = bias2  + (size_t)dir*GDIM;
  float*                gates = gates2 + (size_t)dir*NEDU*GDIM;
  int tid = threadIdx.x;
  for (int i = tid; i < NEDU*HDIM; i += blockDim.x) cF[i] = 0.f;
  for (int i = tid; i < NEDU*KPH;  i += blockDim.x) hB[i] = 0;
  // stage Whh into LDS with the Tensor Data Mover (one wave issues the DMA)
  if (tid < 32) {
    unsigned lds_off = (unsigned)(uintptr_t)(void*)hW;
    tdm_load_whh(Whh, lds_off);
    __builtin_amdgcn_s_wait_tensorcnt(0);
  }
  __syncthreads();
  int wave = tid >> 5, lane = tid & 31;
  int lrow = lane & 15, ks = (lane >> 4) << 4;
  for (int step = 0; step < TT; ++step){
    int t = dir ? (TT-1-step) : step;
    // phase 1: gate pre-activations = h@Whh^T + bias + xproj, tiled over waves
    for (int tile = wave; tile < 16*(GDIM/16); tile += 32){
      int m0 = (tile & 15)*16, n0 = (tile >> 4)*16;
      v8f acc = {};
      #pragma unroll
      for (int k0 = 0; k0 < KPH; k0 += 32){
        v16bf a = *reinterpret_cast<const v16bf*>(hB + (size_t)(m0+lrow)*KPH + k0 + ks);
        v16bf b = *reinterpret_cast<const v16bf*>(hW + (size_t)(n0+lrow)*KPH + k0 + ks);
        acc = __builtin_amdgcn_wmma_f32_16x16x32_bf16(false, a, false, b, (short)0, acc, false, false);
      }
      int col = n0 + (lane & 15);
      int rb  = m0 + ((lane >> 4) << 3);
      float bb = bias[col];
      #pragma unroll
      for (int r = 0; r < 8; ++r){
        int brow = rb + r;
        gates[(size_t)brow*GDIM + col] =
            acc[r] + bb + xproj[((size_t)brow*TT + t)*GDIM + col];
      }
    }
    __syncthreads();
    // phase 2: elementwise LSTM cell + length masking (carry lives in cF + hB)
    for (int idx = tid; idx < NEDU*HDIM; idx += blockDim.x){
      int b_ = idx / HDIM, j = idx % HDIM;
      const float* g = gates + (size_t)b_*GDIM;
      float ig = g[j], fg = g[HDIM+j], gg = g[2*HDIM+j], og = g[3*HDIM+j];
      float cold = cF[idx];
      float cn = sigmoidf_(fg)*cold + sigmoidf_(ig)*tanhf(gg);
      float hn = sigmoidf_(og)*tanhf(cn);
      bool valid = t < lengths[b_];
      if (valid){
        cF[idx] = cn;
        hB[(size_t)b_*KPH + j] = f2bf(hn);
      }
      float ho = valid ? hn : 0.0f;
      size_t o = (size_t)b_*TT + t;
      outs[o*NHD + dir*HDIM + j]    = ho;
      nextin[o*KP12 + dir*HDIM + j] = f2bf(ho);
    }
    __syncthreads();
  }
}

// ---------------- masked mean pool over time ---------------------------------
__global__ void k_pool(const float* __restrict__ outs, const int* __restrict__ lengths,
                       float* __restrict__ enc){
  int b = blockIdx.x, j = threadIdx.x;   // blockDim == 200
  float s = 0.f;
  for (int t = 0; t < TT; ++t) s += outs[((size_t)b*TT + t)*NHD + j];
  enc[(size_t)b*NHD + j] = s / (float)lengths[b];
}

// ---------------- sequential shift-reduce parser (single WG, stack in LDS) ---
__global__ void k_parser(const float* __restrict__ enc, const float* __restrict__ missing,
                         const float* __restrict__ Wa, const float* __restrict__ ba,
                         const float* __restrict__ Wt, const float* __restrict__ bt,
                         float* __restrict__ out){
  extern __shared__ char smem[];
  float* stack  = (float*)smem;           // 256*200
  float* feat   = stack + NEDU*NHD;       // 600
  float* gvec   = feat + 600;             // 500
  float* merged = gvec + 500;             // 200
  float* scores = merged + 200;           // 2
  __shared__ int sh_sp, sh_bp, sh_isshift, sh_spn, sh_bpn;
  int tid = threadIdx.x;
  // warm L2/L0 with the 400KB Wt matrix that the serial chain re-reads 511x
  {
    const char* wb = (const char*)Wt;
    for (int i = tid; i < (5*HDIM*NHD*4)/128; i += (int)blockDim.x)
      __builtin_prefetch(wb + (size_t)i*128, 0, 1);
  }
  if (tid == 0){ sh_sp = 0; sh_bp = 0; }
  __syncthreads();
  for (int step = 0; step < 2*NEDU - 1; ++step){
    int sp = sh_sp, bp = sh_bp;
    // P1: build feat = [s1, s0, b]
    for (int i = tid; i < 600; i += blockDim.x){
      int region = i / NHD, j = i % NHD;
      float v;
      if (region == 0)      v = (sp >= 2)   ? stack[(size_t)min(max(sp-2,0),NEDU-1)*NHD + j] : missing[j];
      else if (region == 1) v = (sp >= 1)   ? stack[(size_t)min(max(sp-1,0),NEDU-1)*NHD + j] : missing[j];
      else                  v = (bp < NEDU) ? enc[(size_t)min(bp,NEDU-1)*NHD + j]            : missing[j];
      feat[i] = v;
    }
    __syncthreads();
    // P2: tree-LSTM gate vector (threads 0..499) + action scores (waves 16,17)
    if (tid < 500){
      float acc = bt[tid];
      const float* wr = Wt + (size_t)tid*NHD;
      for (int k = 0; k < HDIM; ++k) acc += wr[k]       * feat[k];        // h1
      for (int k = 0; k < HDIM; ++k) acc += wr[HDIM+k]  * feat[NHD+k];    // h2
      gvec[tid] = acc;
    } else if (tid >= 512 && tid < 576){
      int r = (tid - 512) >> 5;
      int lane = tid & 31;
      float p = 0.f;
      for (int k = lane; k < 600; k += 32) p += Wa[(size_t)r*600 + k] * feat[k];
      for (int off = 16; off > 0; off >>= 1) p += __shfl_down(p, off, 32);
      if (lane == 0) scores[r] = p + ba[r];
    }
    __syncthreads();
    // P3: decide action (tid 0) + merged tree-LSTM output (tids 0..99)
    if (tid == 0){
      const float NEG = -1e30f;
      float s0v = (bp < NEDU) ? scores[0] : NEG;   // shift
      float s1v = (sp >= 2)   ? scores[1] : NEG;   // reduce
      int is_shift = !(s1v > s0v);                 // argmax ties -> index 0
      sh_isshift = is_shift;
      sh_spn = is_shift ? sp + 1 : sp - 1;
      sh_bpn = is_shift ? bp + 1 : bp;
    }
    if (tid < HDIM){
      float i_ = gvec[tid], f1 = gvec[HDIM+tid], f2 = gvec[2*HDIM+tid],
            o_ = gvec[3*HDIM+tid], u_ = gvec[4*HDIM+tid];
      float c1 = feat[HDIM+tid], c2 = feat[NHD+HDIM+tid];
      float c = sigmoidf_(f1)*c1 + sigmoidf_(f2)*c2 + sigmoidf_(i_)*tanhf(u_);
      float h = sigmoidf_(o_)*tanhf(c);
      merged[tid] = h; merged[HDIM+tid] = c;
    }
    __syncthreads();
    // P4: stack update
    if (tid < NHD){
      if (sh_isshift){
        int idx = min(max(sp,0),NEDU-1);
        stack[(size_t)idx*NHD + tid] = feat[2*NHD + tid];
      } else {
        int idx = min(max(sp-2,0),NEDU-1);
        stack[(size_t)idx*NHD + tid] = merged[tid];
      }
    }
    __syncthreads();
    if (tid == 0){ sh_sp = sh_spn; sh_bp = sh_bpn; }
    __syncthreads();
  }
  if (tid < NHD) out[tid] = stack[tid];
}

extern "C" void kernel_launch(void* const* d_in, const int* in_sizes, int n_in,
                              void* d_out, int out_size, void* d_ws, size_t ws_size,
                              hipStream_t stream){
  const int*   tokens  = (const int*)d_in[0];
  const int*   lengths = (const int*)d_in[1];
  const float* emb     = (const float*)d_in[2];
  const float* Wih_l0  = (const float*)d_in[3];
  const float* Whh_l0  = (const float*)d_in[4];
  const float* b_l0    = (const float*)d_in[5];
  const float* Wih_l12 = (const float*)d_in[6];
  const float* Whh_l12 = (const float*)d_in[7];
  const float* b_l12   = (const float*)d_in[8];
  const float* missing = (const float*)d_in[9];
  const float* Wa      = (const float*)d_in[10];
  const float* ba      = (const float*)d_in[11];
  const float* Wt      = (const float*)d_in[12];
  const float* bt      = (const float*)d_in[13];
  float* out = (float*)d_out;
  (void)in_sizes; (void)n_in; (void)out_size; (void)ws_size;

  char* ws = (char*)d_ws;
  size_t off = 0;
  auto alloc = [&](size_t bytes)->char*{
    char* p = ws + off; off += (bytes + 255) & ~size_t(255); return p;
  };
  unsigned short* we     = (unsigned short*)alloc((size_t)MROWS*KP0*2);
  unsigned short* inb    = (unsigned short*)alloc((size_t)MROWS*KP12*2);
  float*          xproj  = (float*)         alloc(2ull*MROWS*GDIM*4);
  unsigned short* wih0b  = (unsigned short*)alloc(2ull*GDIM*KP0*2);
  unsigned short* wih12b = (unsigned short*)alloc(4ull*GDIM*KP12*2);
  unsigned short* whh0b  = (unsigned short*)alloc(2ull*GDIM*KPH*2);
  unsigned short* whh12b = (unsigned short*)alloc(4ull*GDIM*KPH*2);
  float*          gates  = (float*)         alloc(2ull*NEDU*GDIM*4);
  float*          outs   = (float*)         alloc((size_t)NEDU*TT*NHD*4);
  float*          enc    = (float*)         alloc((size_t)NEDU*NHD*4);

  hipLaunchKernelGGL(k_embed, dim3(MROWS), dim3(KP0), 0, stream, tokens, emb, we);
  hipLaunchKernelGGL(k_cvt, dim3(2*GDIM), dim3(KP0),  0, stream, Wih_l0,  wih0b,  EDIM, KP0);
  hipLaunchKernelGGL(k_cvt, dim3(4*GDIM), dim3(KP12), 0, stream, Wih_l12, wih12b, NHD,  KP12);
  hipLaunchKernelGGL(k_cvt, dim3(2*GDIM), dim3(KPH),  0, stream, Whh_l0,  whh0b,  HDIM, KPH);
  hipLaunchKernelGGL(k_cvt, dim3(4*GDIM), dim3(KPH),  0, stream, Whh_l12, whh12b, HDIM, KPH);
  {
    int n = MROWS*(KP12-NHD);
    hipLaunchKernelGGL(k_zpad, dim3((n+255)/256), dim3(256), 0, stream, inb);
  }
  for (int L = 0; L < 3; ++L){
    const unsigned short* A  = (L==0) ? we : inb;
    int Kp = (L==0) ? KP0 : KP12;
    const unsigned short* Bw = (L==0) ? wih0b : (wih12b + (size_t)(L-1)*2*GDIM*KP12);
    const unsigned short* Wh = (L==0) ? whh0b : (whh12b + (size_t)(L-1)*2*GDIM*KPH);
    const float* bias        = (L==0) ? b_l0  : (b_l12  + (size_t)(L-1)*2*GDIM);
    hipLaunchKernelGGL(k_gemm_xproj, dim3(1600,2), dim3(256), 0, stream, A, Bw, xproj, Kp);
    size_t scanLds = (size_t)NEDU*HDIM*4 + (size_t)NEDU*KPH*2 + (size_t)GDIM*KPH*2; // 270336
    hipLaunchKernelGGL(k_scan, dim3(2), dim3(1024), scanLds, stream,
                       xproj, Wh, bias, lengths, outs, inb, gates);
  }
  hipLaunchKernelGGL(k_pool, dim3(NEDU), dim3(NHD), 0, stream, outs, lengths, enc);
  size_t parLds = ((size_t)NEDU*NHD + 600 + 500 + 200 + 2)*4;       // ~206 KB
  hipLaunchKernelGGL(k_parser, dim3(1), dim3(1024), parLds, stream,
                     enc, missing, Wa, ba, Wt, bt, out);
}
